// LocalRNN_84619445666139
// MI455X (gfx1250) — matrix operand order repaired and checked by reading, not compiled, not run on hardware
//
#include <hip/hip_runtime.h>

typedef __attribute__((ext_vector_type(16))) _Float16 v16h;
typedef __attribute__((ext_vector_type(8)))  _Float16 v8h;
typedef __attribute__((ext_vector_type(8)))  float    v8f;
typedef __attribute__((ext_vector_type(4)))  float    v4f;

#define DEV __device__ __forceinline__

// Problem constants (from setup_inputs)
constexpr int CB = 16;            // batch
constexpr int CL = 1024;          // seq len
constexpr int CD = 256;           // input dim
constexpr int CH = 256;           // hidden dim
constexpr int CK = 16;            // window
constexpr int JR = CL + CK - 1;   // 1039 xpad rows per batch
constexpr int NG = 3 * CH;        // 768 gate columns
constexpr int RTOT = CB * JR;     // 16624 flattened xpad rows (multiple of 16)
constexpr int GI_ROWS_LDS = 2 * CK - 1;   // 31 Gi rows per l-tile

DEV float sigf(float x) { return 1.0f / (1.0f + __expf(-x)); }
DEV float tanhfast(float x) { return 2.0f / (1.0f + __expf(-2.0f * x)) - 1.0f; }

// Load 16 consecutive f32 and convert to a v16h (used for B fragments:
// element i <-> K = base + i, with base = kc*32 + 16*(lane>=16)).
DEV v16h load_cvt16(const float* __restrict__ p) {
  union { v16h v; _Float16 e[16]; } u;
  v4f a = *(const v4f*)(p);
  v4f b = *(const v4f*)(p + 4);
  v4f c = *(const v4f*)(p + 8);
  v4f d = *(const v4f*)(p + 12);
#pragma unroll
  for (int i = 0; i < 4; ++i) {
    u.e[i]      = (_Float16)a[i];
    u.e[4 + i]  = (_Float16)b[i];
    u.e[8 + i]  = (_Float16)c[i];
    u.e[12 + i] = (_Float16)d[i];
  }
  return u.v;
}

// ---------------------------------------------------------------------------
// Kernel 1: Gi[b, j, 0:768] = xpad[b,j,:] @ W_ih^T + b_ih (+ b_hh for r,z)
// One wave per 16x16 output tile. M = flattened xpad row, K = D = 256.
// Stored as f16 in workspace.
// ---------------------------------------------------------------------------
__global__ __launch_bounds__(128) void gi_gemm_kernel(
    const float* __restrict__ x, const float* __restrict__ Wih,
    const float* __restrict__ bih, const float* __restrict__ bhh,
    _Float16* __restrict__ Gi) {
  const int lane = threadIdx.x & 31;
  const int wv   = threadIdx.x >> 5;
  const int lh   = lane >> 4;      // half (0/1)
  const int lc   = lane & 15;
  const int tileR = blockIdx.x;                 // 0..1038
  const int ntile = blockIdx.y * 4 + wv;        // 0..47
  const int n0    = ntile * 16;

  // This lane's A row for loading (A map: M = lane&15)
  const int R = tileR * 16 + lc;                // flattened xpad row
  const int b = R / JR;
  const int j = R - b * JR;
  const bool pad = (j < CK - 1);
  const float* xrow = x + ((size_t)b * CL + (j - (CK - 1))) * (size_t)CD;

  v8f acc = {};
#pragma unroll
  for (int kc = 0; kc < 8; ++kc) {
    // A fragment: elements 0..15 -> K = kb..kb+15
    union { v16h v; _Float16 e[16]; } A;
    const int kb = kc * 32 + 16 * lh;
    if (!pad) {
      v4f r0 = *(const v4f*)(xrow + kb);
      v4f r1 = *(const v4f*)(xrow + kb + 4);
      v4f r2 = *(const v4f*)(xrow + kb + 8);
      v4f r3 = *(const v4f*)(xrow + kb + 12);
#pragma unroll
      for (int i = 0; i < 4; ++i) {
        A.e[i]      = (_Float16)r0[i];
        A.e[4 + i]  = (_Float16)r1[i];
        A.e[8 + i]  = (_Float16)r2[i];
        A.e[12 + i] = (_Float16)r3[i];
      }
    } else {
#pragma unroll
      for (int i = 0; i < 16; ++i) A.e[i] = (_Float16)0.0f;
    }
    // B fragment from W_ih[n][k] (row-major [768][256]): N = n0+lc
    const float* wrow = Wih + (size_t)(n0 + lc) * CD + kc * 32 + 16 * lh;
    v16h Bf = load_cvt16(wrow);
    acc = __builtin_amdgcn_wmma_f32_16x16x32_f16(false, A.v, false, Bf,
                                                 (short)0, acc, false, false);
  }

  // Fold biases: r,z gates get b_ih + b_hh ; n gate gets b_ih only.
  const int ncol = n0 + lc;
  float bias = bih[ncol] + ((ntile < 32) ? bhh[ncol] : 0.0f);

#pragma unroll
  for (int r = 0; r < 8; ++r) {
    const int m = r + 8 * lh;                    // C/D map
    Gi[(size_t)(tileR * 16 + m) * NG + ncol] = (_Float16)(acc[r] + bias);
  }
}

// ---------------------------------------------------------------------------
// Kernel 2: the 16-step GRU recurrence for a (b, 16-row l-tile).
// 16 waves; wave w owns H-columns [16w, 16w+16) of gates r,z,n.
// h (f16) double-buffered in LDS; W_hh fragments register-resident.
// Gi staged into LDS with GLOBAL_LOAD_ASYNC_TO_LDS_B128 (ASYNCcnt path).
// ---------------------------------------------------------------------------
__global__ __launch_bounds__(512) void localrnn_kernel(
    const float* __restrict__ Whh, const float* __restrict__ bhh,
    const _Float16* __restrict__ Gi, float* __restrict__ out) {
  __shared__ __align__(16) _Float16 sh_h[2][16][CH];            // 16 KB
  __shared__ __align__(16) _Float16 sh_gi[GI_ROWS_LDS * NG];    // 46.5 KB

  const int tid  = threadIdx.x;
  const int lane = tid & 31;
  const int wv   = tid >> 5;       // 0..15
  const int lh   = lane >> 4;
  const int lc   = lane & 15;
  const int l0   = blockIdx.x * 16;
  const int b    = blockIdx.y;
  const int c    = wv * 16;        // column offset within H

  // Stage Gi rows [b*JR + l0 .. +30] into LDS via async DMA (no VGPR data).
  // 31*768 f16 = 47616 B = 2976 x 16B chunks; GVS addressing: saddr = base,
  // vaddr = per-lane byte offset, vdst = LDS byte address.
  {
    const unsigned long long gbase =
        (unsigned long long)(const void*)(Gi + ((size_t)b * JR + l0) * (size_t)NG);
    const unsigned lds_base = (unsigned)(unsigned long long)(const void*)sh_gi;
    constexpr int N4 = GI_ROWS_LDS * NG / 8;   // 2976 chunks of 16 B
    for (int idx = tid; idx < N4; idx += 512) {
      const unsigned off = (unsigned)idx * 16u;
      const unsigned ldsa = lds_base + off;
      asm volatile("global_load_async_to_lds_b128 %0, %1, %2"
                   :: "v"(ldsa), "v"(off), "s"(gbase)
                   : "memory");
    }
  }
  // Zero h buffer 0 (8192 B = 512 uint4, one per thread).
  {
    uint4 z; z.x = z.y = z.z = z.w = 0u;
    ((uint4*)&sh_h[0][0][0])[tid] = z;
  }

  // Preload W_hh fragments: gate g, k-chunk kc; row n = g*256 + c + lc.
  v16h Bf[3][8];
#pragma unroll
  for (int g = 0; g < 3; ++g)
#pragma unroll
    for (int kc = 0; kc < 8; ++kc)
      Bf[g][kc] = load_cvt16(Whh + (size_t)(g * CH + c + lc) * CH +
                             kc * 32 + 16 * lh);

  const float bn = bhh[2 * CH + c + lc];   // b_hh for n-gate (inside r*(...))

  float h[8];
#pragma unroll
  for (int r = 0; r < 8; ++r) h[r] = 0.0f;

  // All async Gi->LDS transfers of this wave must land before the barrier.
  asm volatile("s_wait_asynccnt 0" ::: "memory");
  __syncthreads();

#pragma unroll 1
  for (int t = 0; t < CK; ++t) {
    const int rb = t & 1, wb = rb ^ 1;
    v8f accR = {}, accZ = {}, accN = {};
#pragma unroll
    for (int kc = 0; kc < 8; ++kc) {
      // A fragment of h: row = lc, K runs [kb,kb+16)
      union { v16h v; v8h p[2]; } A;
      const int kb = kc * 32 + 16 * lh;
      A.p[0] = *(const v8h*)&sh_h[rb][lc][kb];
      A.p[1] = *(const v8h*)&sh_h[rb][lc][kb + 8];
      accR = __builtin_amdgcn_wmma_f32_16x16x32_f16(false, A.v, false, Bf[0][kc],
                                                    (short)0, accR, false, false);
      accZ = __builtin_amdgcn_wmma_f32_16x16x32_f16(false, A.v, false, Bf[1][kc],
                                                    (short)0, accZ, false, false);
      accN = __builtin_amdgcn_wmma_f32_16x16x32_f16(false, A.v, false, Bf[2][kc],
                                                    (short)0, accN, false, false);
    }
    // GRU elementwise update for this wave's 8 (row, col) elements.
#pragma unroll
    for (int r = 0; r < 8; ++r) {
      const int m = r + 8 * lh;
      const int gbase = (m + t) * NG + c + lc;
      float gir = (float)sh_gi[gbase];
      float giz = (float)sh_gi[gbase + CH];
      float gin = (float)sh_gi[gbase + 2 * CH];
      float rr = sigf(gir + accR[r]);
      float zz = sigf(giz + accZ[r]);
      float nn = tanhfast(gin + rr * (accN[r] + bn));
      h[r] = (1.0f - zz) * nn + zz * h[r];
      sh_h[wb][m][c + lc] = (_Float16)h[r];
    }
    __syncthreads();
  }

  // Write final hidden states: out[b, l0+m, c+lc]
#pragma unroll
  for (int r = 0; r < 8; ++r) {
    const int m = r + 8 * lh;
    out[((size_t)b * CL + l0 + m) * (size_t)CH + c + lc] = h[r];
  }
}

// ---------------------------------------------------------------------------
extern "C" void kernel_launch(void* const* d_in, const int* in_sizes, int n_in,
                              void* d_out, int out_size, void* d_ws, size_t ws_size,
                              hipStream_t stream) {
  const float* x    = (const float*)d_in[0];   // [16,1024,256]
  const float* Wih  = (const float*)d_in[1];   // [768,256]
  const float* Whh  = (const float*)d_in[2];   // [768,256]
  const float* bih  = (const float*)d_in[3];   // [768]
  const float* bhh  = (const float*)d_in[4];   // [768]
  float* out = (float*)d_out;                  // [16,1024,256]
  _Float16* Gi = (_Float16*)d_ws;              // [16624, 768] f16 (~24.4 MB)

  (void)in_sizes; (void)n_in; (void)out_size; (void)ws_size;

  dim3 g1(RTOT / 16, 12);        // 1039 M-tiles x (12 blocks * 4 waves = 48 N-tiles)
  gi_gemm_kernel<<<g1, 128, 0, stream>>>(x, Wih, bih, bhh, Gi);

  dim3 g2(CL / 16, CB);          // 64 l-tiles x 16 batches
  localrnn_kernel<<<g2, 512, 0, stream>>>(Whh, bhh, Gi, out);
}
